// Mamba2DBlock_9749575762048
// MI455X (gfx1250) — matrix operand order repaired and checked by reading, not compiled, not run on hardware
//
#include <hip/hip_runtime.h>
#include <hip/hip_bf16.h>

typedef __attribute__((ext_vector_type(16))) _Float16 v16h;
typedef __attribute__((ext_vector_type(8)))  _Float16 v8h;
typedef __attribute__((ext_vector_type(8)))  float    v8f;

#define B_SZ    8
#define C_CH    512
#define HH      24
#define WW      24
#define L_SEQ   576          // HH*WW
#define D_INNER 1024
#define D_STATE 16
#define DT_RANK 32
#define N_TOK   (B_SZ * L_SEQ)   // 4608

// ---------------------------------------------------------------------------
// f32 -> f16 elementwise (weight conversion, done once per launch)
// ---------------------------------------------------------------------------
__global__ __launch_bounds__(256) void f32_to_f16_k(const float* __restrict__ in,
                                                    _Float16* __restrict__ out, int n) {
    int i = blockIdx.x * 256 + threadIdx.x;
    if (i < n) out[i] = (_Float16)in[i];
}

// ---------------------------------------------------------------------------
// LayerNorm over C=512: gather (B,C,H,W)-strided channels for token (b,l),
// write f16 row-major (token, C) ready to be WMMA A-matrix.
// ---------------------------------------------------------------------------
__global__ __launch_bounds__(256) void layernorm_k(const float* __restrict__ x,
                                                   const float* __restrict__ g,
                                                   const float* __restrict__ beta,
                                                   _Float16* __restrict__ xn) {
    __shared__ float red[256];
    int m = blockIdx.x;
    int b = m / L_SEQ, l = m - b * L_SEQ;
    const float* base = x + (size_t)b * C_CH * L_SEQ + l;
    int t = threadIdx.x;
    float v0 = base[(size_t)t * L_SEQ];
    float v1 = base[(size_t)(t + 256) * L_SEQ];
    red[t] = v0 + v1;
    __syncthreads();
    for (int s = 128; s > 0; s >>= 1) { if (t < s) red[t] += red[t + s]; __syncthreads(); }
    float mu = red[0] * (1.0f / C_CH);
    __syncthreads();
    float d0 = v0 - mu, d1 = v1 - mu;
    red[t] = d0 * d0 + d1 * d1;
    __syncthreads();
    for (int s = 128; s > 0; s >>= 1) { if (t < s) red[t] += red[t + s]; __syncthreads(); }
    float rs = rsqrtf(red[0] * (1.0f / C_CH) + 1e-5f);
    _Float16* o = xn + (size_t)m * C_CH;
    o[t]       = (_Float16)(d0 * rs * g[t] + beta[t]);
    o[t + 256] = (_Float16)(d1 * rs * g[t + 256] + beta[t + 256]);
}

// ---------------------------------------------------------------------------
// Generic WMMA GEMM: D[M,N] (f32) = A[M,K](f16, row-major) * W[N,K](f16)^T
// One wave computes a 16x64 tile via 4x v_wmma_f32_16x16x32_f16 per K-step.
// EPI 0: plain f32 store
// EPI 1: f32 store + f16 dual-store of cols < DT_RANK (dt slice of x_dbl)
// EPI 2: fused bias + softplus (dt projection)
// EPI 3: fused residual add + (B,L,C)->(B,C,H,W) scatter into d_out
// ---------------------------------------------------------------------------
template <int EPI>
__global__ __launch_bounds__(256) void gemm_wmma_k(
    const _Float16* __restrict__ A, const _Float16* __restrict__ W,
    float* __restrict__ Cout, int M, int N, int K,
    const float* __restrict__ bias, _Float16* __restrict__ out16,
    const float* __restrict__ resid, float* __restrict__ outimg) {
    int wv = blockIdx.x * 8 + (threadIdx.x >> 5);
    int ntn = N >> 6;
    int total = (M >> 4) * ntn;
    if (wv >= total) return;
    int tm = wv / ntn;
    int tn = wv - tm * ntn;
    int m0 = tm << 4, n0 = tn << 6;
    int lane = threadIdx.x & 31;
    int r = lane & 15, hi = lane >> 4;

    v8f acc[4] = {};
    const _Float16* ap = A + (size_t)(m0 + r) * K + (hi << 4);
    const _Float16* wp = W + (size_t)(n0 + r) * K + (hi << 4);

    for (int k0 = 0; k0 < K; k0 += 32) {
        v16h a = *(const v16h*)(ap + k0);
#pragma unroll
        for (int j = 0; j < 4; ++j) {
            v16h bm = *(const v16h*)(wp + (size_t)j * 16 * K + k0);
            acc[j] = __builtin_amdgcn_wmma_f32_16x16x32_f16(
                false, a, false, bm, (short)0, acc[j], false, false);
        }
    }

#pragma unroll
    for (int j = 0; j < 4; ++j) {
        int col = n0 + j * 16 + r;
#pragma unroll
        for (int i = 0; i < 8; ++i) {
            int row = m0 + i + (hi << 3);
            float v = acc[j][i];
            if (EPI == 0) {
                Cout[(size_t)row * N + col] = v;
            } else if (EPI == 1) {
                Cout[(size_t)row * N + col] = v;
                if (col < DT_RANK) out16[(size_t)row * DT_RANK + col] = (_Float16)v;
            } else if (EPI == 2) {
                v += bias[col];
                v = (v > 20.f) ? v : log1pf(__expf(v));
                Cout[(size_t)row * N + col] = v;
            } else { // EPI == 3: residual + scatter to (B,C,H,W)
                int bb = row / L_SEQ;
                int l  = row - bb * L_SEQ;
                size_t idx = ((size_t)bb * C_CH + col) * L_SEQ + l;
                outimg[idx] = resid[idx] + v;
            }
        }
    }
}

// ---------------------------------------------------------------------------
// Causal depthwise conv1d (k=3) over L + bias + SiLU; emits f32 and f16 copies.
// ---------------------------------------------------------------------------
__global__ __launch_bounds__(256) void conv_silu_k(const float* __restrict__ xz,
                                                   const float* __restrict__ cw,
                                                   const float* __restrict__ cb,
                                                   float* __restrict__ xc,
                                                   _Float16* __restrict__ xc16) {
    int i = blockIdx.x * 256 + threadIdx.x;
    if (i >= N_TOK * D_INNER) return;
    int d = i & (D_INNER - 1);
    int tok = i >> 10;
    int b = tok / L_SEQ, l = tok - b * L_SEQ;
    (void)b;
    const float* xi = xz + (size_t)tok * (2 * D_INNER) + d;
    float v = cw[d * 3 + 2] * xi[0];
    if (l >= 1) v += cw[d * 3 + 1] * xi[-(ptrdiff_t)(2 * D_INNER)];
    if (l >= 2) v += cw[d * 3 + 0] * xi[-(ptrdiff_t)(4 * D_INNER)];
    v += cb[d];
    v = v / (1.0f + __expf(-v));       // SiLU
    xc[i]   = v;
    xc16[i] = (_Float16)v;
}

// ---------------------------------------------------------------------------
// Selective scan + skip + SiLU(z) gate, fused, f16 output for out_proj.
// grid (B, 4), block 256: thread owns one d channel (h[16] in registers);
// B/C vectors (shared across all d) staged in LDS per 48-step chunk.
// ---------------------------------------------------------------------------
#define SCAN_CHUNK 48
__global__ __launch_bounds__(256) void scan_gate_k(const float* __restrict__ xdbl,
                                                   const float* __restrict__ dt,
                                                   const float* __restrict__ xc,
                                                   const float* __restrict__ xz,
                                                   const float* __restrict__ Alog,
                                                   const float* __restrict__ Dp,
                                                   _Float16* __restrict__ y16) {
    __shared__ float sBC[SCAN_CHUNK * 32];
    int b = blockIdx.x;
    int d = blockIdx.y * 256 + threadIdx.x;

    float Ad[D_STATE];
#pragma unroll
    for (int s = 0; s < D_STATE; ++s) Ad[s] = -__expf(Alog[d * D_STATE + s]);
    float Dd = Dp[d];
    float h[D_STATE] = {};

    for (int c = 0; c < L_SEQ / SCAN_CHUNK; ++c) {
        int l0 = c * SCAN_CHUNK;
        for (int i = threadIdx.x; i < SCAN_CHUNK * 32; i += 256) {
            int ll = i >> 5, t = i & 31;
            sBC[i] = xdbl[((size_t)(b * L_SEQ + l0 + ll)) * 64 + DT_RANK + t];
        }
        __syncthreads();
        for (int ll = 0; ll < SCAN_CHUNK; ++ll) {
            size_t tok = (size_t)b * L_SEQ + l0 + ll;
            float dtv = dt[tok * D_INNER + d];
            float xv  = xc[tok * D_INNER + d];
            float zv  = xz[tok * (2 * D_INNER) + D_INNER + d];
            const float* bcp = &sBC[ll * 32];
            float y = 0.f;
#pragma unroll
            for (int s = 0; s < D_STATE; ++s) {
                float dA = __expf(dtv * Ad[s]);
                h[s] = dA * h[s] + dtv * bcp[s] * xv;
                y += h[s] * bcp[D_STATE + s];
            }
            y += Dd * xv;
            y *= zv / (1.0f + __expf(-zv));   // * SiLU(z)
            y16[tok * D_INNER + d] = (_Float16)y;
        }
        __syncthreads();
    }
}

// ---------------------------------------------------------------------------
// Launch
// ---------------------------------------------------------------------------
extern "C" void kernel_launch(void* const* d_in, const int* in_sizes, int n_in,
                              void* d_out, int out_size, void* d_ws, size_t ws_size,
                              hipStream_t stream) {
    const float* x      = (const float*)d_in[0];
    const float* ln_g   = (const float*)d_in[1];
    const float* ln_b   = (const float*)d_in[2];
    const float* in_w   = (const float*)d_in[3];
    const float* conv_w = (const float*)d_in[4];
    const float* conv_b = (const float*)d_in[5];
    const float* xp_w   = (const float*)d_in[6];
    const float* dt_w   = (const float*)d_in[7];
    const float* dt_b   = (const float*)d_in[8];
    const float* A_log  = (const float*)d_in[9];
    const float* D_par  = (const float*)d_in[10];
    const float* out_w  = (const float*)d_in[11];
    float* out = (float*)d_out;

    char* ws = (char*)d_ws;
    size_t off = 0;
    auto alloc = [&](size_t bytes) {
        size_t o = off;
        off += (bytes + 255) & ~(size_t)255;
        return o;
    };
    float*    xz   = (float*)(ws + alloc((size_t)N_TOK * 2 * D_INNER * 4));
    float*    xc   = (float*)(ws + alloc((size_t)N_TOK * D_INNER * 4));
    float*    xdbl = (float*)(ws + alloc((size_t)N_TOK * 64 * 4));
    float*    dt   = (float*)(ws + alloc((size_t)N_TOK * D_INNER * 4));
    _Float16* xn   = (_Float16*)(ws + alloc((size_t)N_TOK * C_CH * 2));
    _Float16* xc16 = (_Float16*)(ws + alloc((size_t)N_TOK * D_INNER * 2));
    _Float16* dtin = (_Float16*)(ws + alloc((size_t)N_TOK * DT_RANK * 2));
    _Float16* y16  = (_Float16*)(ws + alloc((size_t)N_TOK * D_INNER * 2));
    _Float16* w_in  = (_Float16*)(ws + alloc((size_t)2 * D_INNER * C_CH * 2));
    _Float16* w_xp  = (_Float16*)(ws + alloc((size_t)64 * D_INNER * 2));
    _Float16* w_dt  = (_Float16*)(ws + alloc((size_t)D_INNER * DT_RANK * 2));
    _Float16* w_out = (_Float16*)(ws + alloc((size_t)C_CH * D_INNER * 2));

    // Weight f32 -> f16 conversions (cheap, once per launch)
    int n;
    n = 2 * D_INNER * C_CH;  f32_to_f16_k<<<(n + 255) / 256, 256, 0, stream>>>(in_w, w_in, n);
    n = 64 * D_INNER;        f32_to_f16_k<<<(n + 255) / 256, 256, 0, stream>>>(xp_w, w_xp, n);
    n = D_INNER * DT_RANK;   f32_to_f16_k<<<(n + 255) / 256, 256, 0, stream>>>(dt_w, w_dt, n);
    n = C_CH * D_INNER;      f32_to_f16_k<<<(n + 255) / 256, 256, 0, stream>>>(out_w, w_out, n);

    // LayerNorm -> f16 token rows
    layernorm_k<<<N_TOK, 256, 0, stream>>>(x, ln_g, ln_b, xn);

    auto blocks = [](int M, int N) { return ((M >> 4) * (N >> 6) + 7) / 8; };

    // in_proj: xz[4608,2048] = xn * w_in^T
    gemm_wmma_k<0><<<blocks(N_TOK, 2 * D_INNER), 256, 0, stream>>>(
        xn, w_in, xz, N_TOK, 2 * D_INNER, C_CH, nullptr, nullptr, nullptr, nullptr);

    // depthwise causal conv + SiLU
    n = N_TOK * D_INNER;
    conv_silu_k<<<(n + 255) / 256, 256, 0, stream>>>(xz, conv_w, conv_b, xc, xc16);

    // x_proj: x_dbl[4608,64] (f32) + f16 dual-store of dt slice
    gemm_wmma_k<1><<<blocks(N_TOK, 64), 256, 0, stream>>>(
        xc16, w_xp, xdbl, N_TOK, 64, D_INNER, nullptr, dtin, nullptr, nullptr);

    // dt_proj + bias + softplus: dt[4608,1024]
    gemm_wmma_k<2><<<blocks(N_TOK, D_INNER), 256, 0, stream>>>(
        dtin, w_dt, dt, N_TOK, D_INNER, DT_RANK, dt_b, nullptr, nullptr, nullptr);

    // selective scan + skip + gate -> y16
    scan_gate_k<<<dim3(B_SZ, 4), 256, 0, stream>>>(xdbl, dt, xc, xz, A_log, D_par, y16);

    // out_proj + residual + layout scatter -> d_out
    gemm_wmma_k<3><<<blocks(N_TOK, C_CH), 256, 0, stream>>>(
        y16, w_out, nullptr, N_TOK, C_CH, D_INNER, nullptr, nullptr, x, out);
}
